// MeshGraphNet_35184372089417
// MI455X (gfx1250) — matrix operand order repaired and checked by reading, not compiled, not run on hardware
//
#include <hip/hip_runtime.h>

#define NDIM     128
#define DEPTH    6
#define NPASSES  4
#define N_NODES  50000
#define N_EDGES  800000
#define NODE_IN  5
#define LN_EPS   1e-5f

typedef __attribute__((ext_vector_type(16))) _Float16 v16h;
typedef __attribute__((ext_vector_type(8)))  float    v8f;
typedef _Float16 half_t;

// ---------------------------------------------------------------------------
// Pre-swizzle f32 weight matrices [nmats][128][128] (row-major, k-major) into
// per-lane WMMA B-fragment order (f16):
//   dst[ mat, ((kb*8+nt)*32 + lane)*16 + hv ] = W[ kb*32 + (lane>>4)*16 + hv ][ nt*16 + (lane&15) ]
// so GEMM inner loops read contiguous 32B per lane from LDS.
// ---------------------------------------------------------------------------
__global__ void swizzle_weights_kernel(const float* __restrict__ src,
                                       half_t* __restrict__ dst, int nmats) {
    int idx = blockIdx.x * blockDim.x + threadIdx.x;
    if (idx >= nmats * 16384) return;
    int mat  = idx >> 14;
    int f    = idx & 16383;
    int hv   = f & 15;
    int lane = (f >> 4) & 31;
    int nt   = (f >> 9) & 7;
    int kb   = f >> 12;
    int k = kb * 32 + ((lane >> 4) << 4) + hv;
    int n = (nt << 4) + (lane & 15);
    dst[idx] = (half_t)src[(size_t)mat * 16384 + k * NDIM + n];
}

// h = v @ enc_W + enc_b
__global__ void encoder_kernel(const float* __restrict__ v,
                               const float* __restrict__ W,
                               const float* __restrict__ b,
                               float* __restrict__ h) {
    int idx = blockIdx.x * blockDim.x + threadIdx.x;
    if (idx >= N_NODES * NDIM) return;
    int n = idx >> 7, c = idx & 127;
    float acc = b[c];
#pragma unroll
    for (int i = 0; i < NODE_IN; ++i) acc += v[n * NODE_IN + i] * W[i * NDIM + c];
    h[idx] = acc;
}

// e = h[ij0] - h[ij1] (+ e)
__global__ void edge_gather_kernel(const float* __restrict__ h,
                                   const int* __restrict__ ij,
                                   float* __restrict__ e, int add_e) {
    int idx = blockIdx.x * blockDim.x + threadIdx.x;
    if (idx >= N_EDGES * NDIM) return;
    int ed = idx >> 7, c = idx & 127;
    int i0 = ij[ed];
    int i1 = ij[N_EDGES + ed];
    float x = h[i0 * NDIM + c] - h[i1 * NDIM + c];
    if (add_e) x += e[idx];
    e[idx] = x;
}

__global__ void copy4_kernel(const float4* __restrict__ src, float4* __restrict__ dst, int n4) {
    int idx = blockIdx.x * blockDim.x + threadIdx.x;
    if (idx < n4) dst[idx] = src[idx];
}

// agg[ij1[E]] += e[E]   (agg pre-initialized with h, so agg = h + scatter_sum)
__global__ void scatter_add_kernel(const float* __restrict__ e,
                                   const int* __restrict__ ij,
                                   float* __restrict__ agg) {
    int idx = blockIdx.x * blockDim.x + threadIdx.x;
    if (idx >= N_EDGES * NDIM) return;
    int ed = idx >> 7, c = idx & 127;
    atomicAdd(&agg[ij[N_EDGES + ed] * NDIM + c], e[idx]);
}

// out = h @ dec_W + dec_b
__global__ void decoder_kernel(const float* __restrict__ h,
                               const float* __restrict__ W,
                               const float* __restrict__ b,
                               float* __restrict__ out) {
    int idx = blockIdx.x * blockDim.x + threadIdx.x;
    if (idx >= N_NODES * NODE_IN) return;
    int n = idx / NODE_IN, j = idx % NODE_IN;
    float acc = b[j];
#pragma unroll 4
    for (int c = 0; c < NDIM; ++c) acc += h[n * NDIM + c] * W[c * NODE_IN + j];
    out[idx] = acc;
}

// ---------------------------------------------------------------------------
// Fused residual block, in place:
//   x <- x + LN(relu(relu(x@W1+b1)@W2+b2)) * g + bt
// 256 threads = 8 wave32; each wave owns one 16x128 row tile.
// W1/W2 staged once per block in LDS in B-fragment order (f16).
// Dynamic LDS layout:
//   [0      , 32KB)  w1 fragments (16384 f16)
//   [32KB   , 64KB)  w2 fragments (16384 f16)
//   [64KB   ,128KB)  per-wave staging: 8 x 2048 floats (f16 view for X/H1, f32 for H2)
//   [128KB  ,130KB)  params: b1|b2|g|bt (4x128 f32)
//   [130KB  ,131KB)  stats: 8 waves x (16 mu + 16 rsig)
// ---------------------------------------------------------------------------
__global__ __launch_bounds__(256) void resblock_wmma_kernel(
    float* __restrict__ x, int nrows,
    const half_t* __restrict__ wf1, const half_t* __restrict__ wf2,
    const float* __restrict__ b1, const float* __restrict__ b2,
    const float* __restrict__ gg, const float* __restrict__ bt) {
    extern __shared__ char smem[];
    half_t* w1s   = (half_t*)smem;
    half_t* w2s   = w1s + 16384;
    float*  stage = (float*)(w2s + 16384);
    float*  prm   = stage + 8 * 2048;
    float*  stats = prm + 4 * NDIM;

    const int tid  = threadIdx.x;
    const int lane = tid & 31;
    const int wv   = tid >> 5;

    // cooperative stage of weights (f16 fragments) and per-column params
    {
        const uint4* s1 = (const uint4*)wf1;
        const uint4* s2 = (const uint4*)wf2;
        uint4* d1 = (uint4*)w1s;
        uint4* d2 = (uint4*)w2s;
        for (int i = tid; i < 2048; i += 256) { d1[i] = s1[i]; d2[i] = s2[i]; }
        for (int i = tid; i < NDIM; i += 256) {
            prm[i] = b1[i]; prm[NDIM + i] = b2[i];
            prm[2 * NDIM + i] = gg[i]; prm[3 * NDIM + i] = bt[i];
        }
    }

    const int ntiles = nrows >> 4;
    int tile = blockIdx.x * 8 + wv;
    if (tile >= ntiles) tile = ntiles - 1;  // duplicate waves recompute identical data
    float* xtile = x + (size_t)tile * 16 * NDIM;

    half_t* sh = (half_t*)(stage + wv * 2048);  // 16x128 f16 view
    float*  sf = stage + wv * 2048;             // 16x128 f32 view

    // stage X tile as f16 (coalesced float4 reads; lane -> 4 columns)
    const int c0 = lane * 4;
#pragma unroll
    for (int r = 0; r < 16; ++r) {
        float4 xv = *(const float4*)(xtile + r * NDIM + c0);
        sh[r * NDIM + c0 + 0] = (half_t)xv.x;
        sh[r * NDIM + c0 + 1] = (half_t)xv.y;
        sh[r * NDIM + c0 + 2] = (half_t)xv.z;
        sh[r * NDIM + c0 + 3] = (half_t)xv.w;
    }
    __syncthreads();  // weights + X staging visible

    const int arow  = lane & 15;
    const int ag    = lane >> 4;
    const int rbase = ag * 8;

    // ----- GEMM1: H1 = relu(X @ W1 + b1) -----
    v16h afr[4];
#pragma unroll
    for (int kb = 0; kb < 4; ++kb) {
        const half_t* lo = sh + arow * NDIM + kb * 32 + ag * 16;
        const half_t* hi = lo + 8;
#pragma unroll
        for (int t = 0; t < 8; ++t) { afr[kb][t] = lo[t]; afr[kb][8 + t] = hi[t]; }
    }
    v8f acc[8];
#pragma unroll
    for (int nt = 0; nt < 8; ++nt) {
        float bv = prm[nt * 16 + arow];
        v8f c = {bv, bv, bv, bv, bv, bv, bv, bv};
#pragma unroll
        for (int kb = 0; kb < 4; ++kb) {
            v16h bfr = *(const v16h*)(w1s + ((kb * 8 + nt) * 32 + lane) * 16);
            c = __builtin_amdgcn_wmma_f32_16x16x32_f16(false, afr[kb], false, bfr,
                                                       (short)0, c, false, false);
        }
        acc[nt] = c;
    }
    __syncthreads();  // all A-frag reads of X staging done

    // spill relu(H1) as f16 back to staging (C/D layout -> row major)
#pragma unroll
    for (int nt = 0; nt < 8; ++nt) {
        int col = nt * 16 + arow;
#pragma unroll
        for (int r = 0; r < 8; ++r) {
            float v = acc[nt][r];
            sh[(rbase + r) * NDIM + col] = (half_t)(v > 0.f ? v : 0.f);
        }
    }
    __syncthreads();

    // ----- GEMM2: H2 = relu(H1 @ W2 + b2) -----
#pragma unroll
    for (int kb = 0; kb < 4; ++kb) {
        const half_t* lo = sh + arow * NDIM + kb * 32 + ag * 16;
        const half_t* hi = lo + 8;
#pragma unroll
        for (int t = 0; t < 8; ++t) { afr[kb][t] = lo[t]; afr[kb][8 + t] = hi[t]; }
    }
#pragma unroll
    for (int nt = 0; nt < 8; ++nt) {
        float bv = prm[NDIM + nt * 16 + arow];
        v8f c = {bv, bv, bv, bv, bv, bv, bv, bv};
#pragma unroll
        for (int kb = 0; kb < 4; ++kb) {
            v16h bfr = *(const v16h*)(w2s + ((kb * 8 + nt) * 32 + lane) * 16);
            c = __builtin_amdgcn_wmma_f32_16x16x32_f16(false, afr[kb], false, bfr,
                                                       (short)0, c, false, false);
        }
        acc[nt] = c;
    }
    __syncthreads();  // A-frag reads of H1 staging done

    // spill relu(H2) as f32 for LayerNorm
#pragma unroll
    for (int nt = 0; nt < 8; ++nt) {
        int col = nt * 16 + arow;
#pragma unroll
        for (int r = 0; r < 8; ++r) {
            float v = acc[nt][r];
            sf[(rbase + r) * NDIM + col] = (v > 0.f ? v : 0.f);
        }
    }
    __syncthreads();

    // per-row mean / rsqrt(var) : lanes 0..15 each reduce one row
    if (lane < 16) {
        const float* rowp = sf + lane * NDIM;
        float s = 0.f, s2 = 0.f;
#pragma unroll 8
        for (int c = 0; c < NDIM; ++c) { float t = rowp[c]; s += t; s2 += t * t; }
        float mu  = s * (1.f / NDIM);
        float var = s2 * (1.f / NDIM) - mu * mu;
        stats[wv * 32 + lane]      = mu;
        stats[wv * 32 + 16 + lane] = rsqrtf(var + LN_EPS);
    }
    __syncthreads();

    // residual + affine LN, streamed out (coalesced float4)
#pragma unroll
    for (int r = 0; r < 16; ++r) {
        float mu = stats[wv * 32 + r];
        float rs = stats[wv * 32 + 16 + r];
        float4 xv = *(const float4*)(xtile + r * NDIM + c0);
        float4 ov;
        ov.x = xv.x + (sf[r * NDIM + c0 + 0] - mu) * rs * prm[2 * NDIM + c0 + 0] + prm[3 * NDIM + c0 + 0];
        ov.y = xv.y + (sf[r * NDIM + c0 + 1] - mu) * rs * prm[2 * NDIM + c0 + 1] + prm[3 * NDIM + c0 + 1];
        ov.z = xv.z + (sf[r * NDIM + c0 + 2] - mu) * rs * prm[2 * NDIM + c0 + 2] + prm[3 * NDIM + c0 + 2];
        ov.w = xv.w + (sf[r * NDIM + c0 + 3] - mu) * rs * prm[2 * NDIM + c0 + 3] + prm[3 * NDIM + c0 + 3];
        *(float4*)(xtile + r * NDIM + c0) = ov;
    }
}

// ---------------------------------------------------------------------------
extern "C" void kernel_launch(void* const* d_in, const int* in_sizes, int n_in,
                              void* d_out, int out_size, void* d_ws, size_t ws_size,
                              hipStream_t stream) {
    const float* v    = (const float*)d_in[0];
    const int*   ij   = (const int*)d_in[1];
    const float* encW = (const float*)d_in[2];
    const float* encb = (const float*)d_in[3];
    const float* eW1  = (const float*)d_in[4];
    const float* eb1  = (const float*)d_in[5];
    const float* eW2  = (const float*)d_in[6];
    const float* eb2  = (const float*)d_in[7];
    const float* eg   = (const float*)d_in[8];
    const float* ebt  = (const float*)d_in[9];
    const float* nW1  = (const float*)d_in[10];
    const float* nb1  = (const float*)d_in[11];
    const float* nW2  = (const float*)d_in[12];
    const float* nb2  = (const float*)d_in[13];
    const float* ng   = (const float*)d_in[14];
    const float* nbt  = (const float*)d_in[15];
    const float* decW = (const float*)d_in[16];
    const float* decb = (const float*)d_in[17];
    float* out = (float*)d_out;

    // workspace carve-up
    char* ws = (char*)d_ws;
    float* h   = (float*)ws;  ws += (size_t)N_NODES * NDIM * sizeof(float);
    float* agg = (float*)ws;  ws += (size_t)N_NODES * NDIM * sizeof(float);
    float* e   = (float*)ws;  ws += (size_t)N_EDGES * NDIM * sizeof(float);
    half_t* eW1f = (half_t*)ws; ws += (size_t)DEPTH * 16384 * sizeof(half_t);
    half_t* eW2f = (half_t*)ws; ws += (size_t)DEPTH * 16384 * sizeof(half_t);
    half_t* nW1f = (half_t*)ws; ws += (size_t)DEPTH * 16384 * sizeof(half_t);
    half_t* nW2f = (half_t*)ws; ws += (size_t)DEPTH * 16384 * sizeof(half_t);

    const dim3 blk(256);
    const size_t smem = 32768 + 32768 + 65536 + 4 * NDIM * 4 + 8 * 32 * 4;

    // pre-swizzle all weights into WMMA fragment order (f16)
    int swg = (DEPTH * 16384 + 255) / 256;
    swizzle_weights_kernel<<<swg, blk, 0, stream>>>(eW1, eW1f, DEPTH);
    swizzle_weights_kernel<<<swg, blk, 0, stream>>>(eW2, eW2f, DEPTH);
    swizzle_weights_kernel<<<swg, blk, 0, stream>>>(nW1, nW1f, DEPTH);
    swizzle_weights_kernel<<<swg, blk, 0, stream>>>(nW2, nW2f, DEPTH);

    encoder_kernel<<<(N_NODES * NDIM + 255) / 256, blk, 0, stream>>>(v, encW, encb, h);

    const int edgeGrid = (N_EDGES * NDIM + 255) / 256;
    const int edgeRB   = N_EDGES / (16 * 8);            // 6250, exact
    const int nodeRB   = (N_NODES / 16 + 7) / 8;        // 391, last block clamps

    for (int p = 0; p < NPASSES; ++p) {
        edge_gather_kernel<<<edgeGrid, blk, 0, stream>>>(h, ij, e, p > 0);
        for (int d = 0; d < DEPTH; ++d)
            resblock_wmma_kernel<<<edgeRB, blk, smem, stream>>>(
                e, N_EDGES, eW1f + d * 16384, eW2f + d * 16384,
                eb1 + d * NDIM, eb2 + d * NDIM, eg + d * NDIM, ebt + d * NDIM);

        copy4_kernel<<<(N_NODES * NDIM / 4 + 255) / 256, blk, 0, stream>>>(
            (const float4*)h, (float4*)agg, N_NODES * NDIM / 4);
        scatter_add_kernel<<<edgeGrid, blk, 0, stream>>>(e, ij, agg);

        for (int d = 0; d < DEPTH; ++d)
            resblock_wmma_kernel<<<nodeRB, blk, smem, stream>>>(
                agg, N_NODES, nW1f + d * 16384, nW2f + d * 16384,
                nb1 + d * NDIM, nb2 + d * NDIM, ng + d * NDIM, nbt + d * NDIM);

        float* t = h; h = agg; agg = t;  // new h lives in agg's buffer
    }

    decoder_kernel<<<(N_NODES * NODE_IN + 255) / 256, blk, 0, stream>>>(h, decW, decb, out);
}
